// ScannedRNN_72438918414878
// MI455X (gfx1250) — compile-verified, matrix operand order checked by reading, not run on hardware
//
#include <hip/hip_runtime.h>
#include <hip/hip_bf16.h>
#include <math.h>

typedef __attribute__((ext_vector_type(16))) __bf16 v16bf;
typedef __attribute__((ext_vector_type(8)))  float  v8f;

#define T_STEPS 512
#define BATCH   64
#define H       512
#define TB      (T_STEPS * BATCH)   // 32768 rows
#define KT      (H / 32)            // 16 k-tiles of 32
#define NT      (H / 16)            // 32 n-tiles of 16
#define LDSS    520                 // padded f32 row stride for staged tiles

// ---------------------------------------------------------------------------
// CDNA5 16-bit A-matrix fragment layout (ISA 7.12.2):
//   lane L: g = L>>4, m = L&15. VGPR v holds K pair (2 per dword).
//   element e (0..15): K = ((e&8)<<1) + g*8 + (e&7)
// B fragment is the mirror with n = L&15 as the column.
// C/D f32: VGPR j, lane (g,n) -> element (M = j + 8g, N = n).
// ---------------------------------------------------------------------------

__device__ __forceinline__ v16bf build_a_from_lds(const float* __restrict__ lds,
                                                  int m, int g, int kbase) {
  v16bf a;
#pragma unroll
  for (int p = 0; p < 8; ++p) {
    // pair (e=2p, 2p+1) -> consecutive K: k0 = ((p&4)<<2) + g*8 + ((p&3)<<1)
    int k = kbase + ((p & 4) << 2) + g * 8 + ((p & 3) << 1);
    float f0 = lds[m * LDSS + k];
    float f1 = lds[m * LDSS + k + 1];
    a[2 * p]     = (__bf16)f0;
    a[2 * p + 1] = (__bf16)f1;
  }
  return a;
}

__device__ __forceinline__ float sigmoidf(float x) {
  return 1.0f / (1.0f + __expf(-x));
}

// ---------------------------------------------------------------------------
// Pack six HxH f32 weight matrices into bf16 WMMA B-fragment tiles.
// Linear layout: (((w*KT + kt)*NT + nt)*32 + lane)*16 bf16 elements.
// ---------------------------------------------------------------------------
__global__ __launch_bounds__(256)
void pack_weights(const float* __restrict__ W0, const float* __restrict__ W1,
                  const float* __restrict__ W2, const float* __restrict__ W3,
                  const float* __restrict__ W4, const float* __restrict__ W5,
                  __bf16* __restrict__ out) {
  int idx = blockIdx.x * 256 + threadIdx.x;       // one thread per (w,kt,nt,lane)
  const int total = 6 * KT * NT * 32;
  if (idx >= total) return;
  int lane = idx & 31;
  int nt   = (idx >> 5) & 31;
  int kt   = (idx >> 10) & 15;
  int w    = idx >> 14;
  const float* W = (w == 0) ? W0 : (w == 1) ? W1 : (w == 2) ? W2
                 : (w == 3) ? W3 : (w == 4) ? W4 : W5;
  int g = lane >> 4, n = lane & 15;
  __bf16* dst = out + (size_t)idx * 16;
#pragma unroll
  for (int e = 0; e < 16; ++e) {
    int k = kt * 32 + (((e & 8) << 1) + g * 8 + (e & 7));
    dst[e] = (__bf16)W[(size_t)k * H + nt * 16 + n];
  }
}

// ---------------------------------------------------------------------------
// Phase 1: x-projections for all T*B rows: xr/xz/xn = ins @ {Wir,Wiz,Win}.
// Block = one 16-row tile; 8 waves, each owns 4 n-tiles x 3 gates.
// ---------------------------------------------------------------------------
__global__ __launch_bounds__(256)
void xproj_kernel(const float* __restrict__ ins,
                  const __bf16* __restrict__ wpack,
                  float* __restrict__ xbuf) {
  __shared__ float lds[16 * LDSS];
  int rt  = blockIdx.x;                 // row tile 0..2047
  int tid = threadIdx.x;
  for (int i = tid; i < 16 * H; i += 256) {
    int r = i >> 9, c = i & (H - 1);
    lds[r * LDSS + c] = ins[(size_t)(rt * 16 + r) * H + c];
  }
  __syncthreads();

  int wv = tid >> 5;                    // 0..7 -> n-tiles [4wv, 4wv+3]
  int lane = tid & 31;
  int g = lane >> 4, n = lane & 15, m = n;

  v8f zero = {0.f, 0.f, 0.f, 0.f, 0.f, 0.f, 0.f, 0.f};
  v8f acc[3][4];
#pragma unroll
  for (int gi = 0; gi < 3; ++gi)
#pragma unroll
    for (int j = 0; j < 4; ++j) acc[gi][j] = zero;

  for (int kt = 0; kt < KT; ++kt) {
    v16bf a = build_a_from_lds(lds, m, g, kt * 32);
#pragma unroll
    for (int gi = 0; gi < 3; ++gi) {
#pragma unroll
      for (int j = 0; j < 4; ++j) {
        int nt = wv * 4 + j;
        const v16bf* bp = (const v16bf*)(wpack +
            ((((size_t)gi * KT + kt) * NT + nt) * 32 + lane) * 16);
        acc[gi][j] = __builtin_amdgcn_wmma_f32_16x16x32_bf16(
            false, a, false, *bp, (short)0, acc[gi][j], false, false);
      }
    }
  }

#pragma unroll
  for (int gi = 0; gi < 3; ++gi) {
#pragma unroll
    for (int j = 0; j < 4; ++j) {
      int col = (wv * 4 + j) * 16 + n;
#pragma unroll
      for (int jj = 0; jj < 8; ++jj) {
        int row = rt * 16 + jj + 8 * g;
        xbuf[(size_t)gi * TB * H + (size_t)row * H + col] = acc[gi][j][jj];
      }
    }
  }
}

// ---------------------------------------------------------------------------
// Phase 2: persistent recurrent scan. 32 blocks x 128 threads (all resident):
//   block (mt = bid&3, qg = bid>>2); wave wv owns n-tile nt = qg*4+wv.
// Per step: stage post-reset h tile (16x512 f32) in LDS, 3 gate GEMMs over
// K=512 via WMMA bf16, fused GRU elementwise, device-scope barrier.
// ---------------------------------------------------------------------------
__global__ __launch_bounds__(128)
void scan_kernel(const float* __restrict__ h0,
                 const int* __restrict__ resets,
                 const __bf16* __restrict__ wpack,     // gates 3,4,5 = Whr,Whz,Whn
                 const float* __restrict__ bhr,
                 const float* __restrict__ bhz,
                 const float* __restrict__ bhn,
                 const float* __restrict__ xbuf,
                 float* __restrict__ hbufA,
                 float* __restrict__ hbufB,
                 float* __restrict__ d_out,            // [0,B*H) h_final, then ys
                 unsigned* __restrict__ ctr) {
  __shared__ float lds[16 * LDSS];
  int mt  = blockIdx.x & 3;
  int qg  = blockIdx.x >> 2;
  int tid = threadIdx.x;
  int wv  = tid >> 5, lane = tid & 31;
  int g = lane >> 4, n = lane & 15, m = n;
  int nt  = qg * 4 + wv;
  int col = nt * 16 + n;

  float br = bhr[col], bz = bhz[col], bn = bhn[col];
  float* ys = d_out + (size_t)BATCH * H;
  v8f zero = {0.f, 0.f, 0.f, 0.f, 0.f, 0.f, 0.f, 0.f};

#pragma unroll 1
  for (int t = 0; t < T_STEPS; ++t) {
    const float* hin = (t == 0) ? h0 : ((t & 1) ? hbufA : hbufB);
    float*       hout = (t & 1) ? hbufB : hbufA;

    // Stage h (post-reset) for this block's 16 batch rows, coalesced.
    for (int i = tid; i < 16 * H; i += 128) {
      int r = i >> 9, c = i & (H - 1);
      int b = mt * 16 + r;
      float hv = resets[t * BATCH + b] ? 0.0f : hin[b * H + c];
      lds[r * LDSS + c] = hv;
    }
    // Prefetch next step's x-projections for this block's rows into L2/L0.
    if (t + 1 < T_STEPS) {
#pragma unroll
      for (int gi = 0; gi < 3; ++gi) {
        const char* p = (const char*)(xbuf + (size_t)gi * TB * H +
                         ((size_t)(t + 1) * BATCH + mt * 16) * H) + (size_t)tid * 256;
        __builtin_prefetch(p, 0, 1);
      }
    }
    __syncthreads();

    v8f accr = zero, accz = zero, accn = zero;
    for (int kt = 0; kt < KT; ++kt) {
      v16bf a = build_a_from_lds(lds, m, g, kt * 32);
      const v16bf* br_p = (const v16bf*)(wpack + ((((size_t)3 * KT + kt) * NT + nt) * 32 + lane) * 16);
      const v16bf* bz_p = (const v16bf*)(wpack + ((((size_t)4 * KT + kt) * NT + nt) * 32 + lane) * 16);
      const v16bf* bn_p = (const v16bf*)(wpack + ((((size_t)5 * KT + kt) * NT + nt) * 32 + lane) * 16);
      accr = __builtin_amdgcn_wmma_f32_16x16x32_bf16(false, a, false, *br_p, (short)0, accr, false, false);
      accz = __builtin_amdgcn_wmma_f32_16x16x32_bf16(false, a, false, *bz_p, (short)0, accz, false, false);
      accn = __builtin_amdgcn_wmma_f32_16x16x32_bf16(false, a, false, *bn_p, (short)0, accn, false, false);
    }

    // Fused GRU elementwise; C/D layout: element jj -> row M = jj + 8g.
#pragma unroll
    for (int jj = 0; jj < 8; ++jj) {
      int M = jj + 8 * g;
      int b = mt * 16 + M;
      size_t xrow = ((size_t)t * BATCH + b) * H + col;
      float xrv = xbuf[xrow];
      float xzv = xbuf[(size_t)TB * H + xrow];
      float xnv = xbuf[(size_t)2 * TB * H + xrow];
      float rr  = sigmoidf(xrv + accr[jj] + br);
      float zz  = sigmoidf(xzv + accz[jj] + bz);
      float nnv = tanhf(xnv + rr * (accn[jj] + bn));
      float hp  = lds[M * LDSS + col];
      float hn  = (1.0f - zz) * nnv + zz * hp;
      __builtin_nontemporal_store(hn, ys + xrow);
      hout[b * H + col] = hn;
      if (t == T_STEPS - 1) d_out[b * H + col] = hn;
    }

    // Device-scope grid barrier: release h writes, wait for all 32 blocks.
    __threadfence();
    __syncthreads();
    if (tid == 0) {
      __hip_atomic_fetch_add(ctr, 1u, __ATOMIC_RELEASE, __HIP_MEMORY_SCOPE_AGENT);
      unsigned target = (unsigned)(t + 1) * 32u;
      while (__hip_atomic_load(ctr, __ATOMIC_ACQUIRE, __HIP_MEMORY_SCOPE_AGENT) < target) {
        __builtin_amdgcn_s_sleep(2);
      }
    }
    __syncthreads();
  }
}

// ---------------------------------------------------------------------------
extern "C" void kernel_launch(void* const* d_in, const int* in_sizes, int n_in,
                              void* d_out, int out_size, void* d_ws, size_t ws_size,
                              hipStream_t stream) {
  const float* ins    = (const float*)d_in[0];
  const int*   resets = (const int*)d_in[1];
  const float* h0     = (const float*)d_in[2];
  const float* Wir    = (const float*)d_in[3];
  const float* Wiz    = (const float*)d_in[4];
  const float* Win    = (const float*)d_in[5];
  const float* Whr    = (const float*)d_in[6];
  const float* Whz    = (const float*)d_in[7];
  const float* Whn    = (const float*)d_in[8];
  const float* bhr    = (const float*)d_in[9];
  const float* bhz    = (const float*)d_in[10];
  const float* bhn    = (const float*)d_in[11];

  // Workspace layout (all 256-aligned):
  //   [ctr 256B][hbufA 128KB][hbufB 128KB][wpack 3MB bf16][xbuf 192MB f32]
  char* ws = (char*)d_ws;
  unsigned* ctr   = (unsigned*)ws;
  float*    hbufA = (float*)(ws + 256);
  float*    hbufB = (float*)(ws + 256 + 131072);
  __bf16*   wpack = (__bf16*)(ws + 256 + 2 * 131072);
  float*    xbuf  = (float*)(ws + 256 + 2 * 131072 + 6u * KT * NT * 32 * 16 * 2);

  hipMemsetAsync(ctr, 0, 256, stream);
  pack_weights<<<(6 * KT * NT * 32 + 255) / 256, 256, 0, stream>>>(
      Wir, Wiz, Win, Whr, Whz, Whn, wpack);
  xproj_kernel<<<TB / 16, 256, 0, stream>>>(ins, wpack, xbuf);
  scan_kernel<<<32, 128, 0, stream>>>(h0, resets, wpack, bhr, bhz, bhn,
                                      xbuf, hbufA, hbufB, (float*)d_out, ctr);
}
